// GBMS_RNN_137438953906
// MI455X (gfx1250) — matrix operand order repaired and checked by reading, not compiled, not run on hardware
//
#include <hip/hip_runtime.h>

// Sizes fixed by the reference
#define BATCH 8
#define NN    4096
#define DD    128
#define ROWS_BLK 128   // rows of Y per workgroup (8 waves x 16 rows)
#define JBLK  64       // columns of W processed per LDS stage
#define LDSTR 132      // padded row stride (ushorts) for row-major LDS planes
#define LDSTRT 68      // padded row stride (ushorts) for transposed LDS plane

typedef __attribute__((ext_vector_type(16))) __bf16 v16bf;
typedef __attribute__((ext_vector_type(8)))  float  v8f;

union Frag { v16bf v; unsigned int u[8]; };

static __device__ inline unsigned int pack2(__bf16 a, __bf16 b) {
  unsigned short ua = __builtin_bit_cast(unsigned short, a);
  unsigned short ub = __builtin_bit_cast(unsigned short, b);
  return (unsigned int)ua | ((unsigned int)ub << 16);
}

// Packed f32->bf16 pair conversion (V_CVT_PK_BF16_F32, verified present)
#if __has_builtin(__builtin_amdgcn_cvt_pk_bf16_f32)
typedef __attribute__((ext_vector_type(2))) __bf16 v2bf;
static __device__ inline unsigned int cvtpk(float a, float b) {
  v2bf r = __builtin_amdgcn_cvt_pk_bf16_f32(a, b);
  return __builtin_bit_cast(unsigned int, r);
}
#else
static __device__ inline unsigned int cvtpk(float a, float b) {
  return pack2((__bf16)a, (__bf16)b);
}
#endif

// Split two f32 into packed bf16 hi + bf16 lo words; hi floats recovered with
// integer ops only (bf16 -> f32 is a 16-bit shift).
static __device__ inline void split_pair(float x0, float x1,
                                         unsigned int& hi, unsigned int& lo) {
  hi = cvtpk(x0, x1);
  const float h0 = __builtin_bit_cast(float, hi << 16);
  const float h1 = __builtin_bit_cast(float, hi & 0xFFFF0000u);
  lo = cvtpk(x0 - h0, x1 - h1);
}

__global__ __launch_bounds__(256)
void gbms_fused(const float* __restrict__ X, const float* __restrict__ bwp,
                float* __restrict__ Y)
{
  // LDS staging of the current 64-row j-block of X:
  //   row-major bf16 hi/lo planes (GEMM1 A-fragments: contiguous K pairs)
  //   transposed bf16 hi plane    (GEMM2 B-fragments: contiguous K pairs)
  __shared__ unsigned short shHi[JBLK * LDSTR];
  __shared__ unsigned short shLo[JBLK * LDSTR];
  __shared__ unsigned short shTHi[DD * LDSTRT];

  const int tid  = threadIdx.x;
  const int lane = tid & 31;
  const int wave = tid >> 5;
  const int b    = blockIdx.x >> 5;            // 32 row-blocks per batch
  const int m0   = (blockIdx.x & 31) * ROWS_BLK;
  const int mw   = m0 + wave * 16;             // this wave's 16 output rows

  const int lm = lane & 15;
  const int kh = lane >> 4;

  const float bw  = bwp[0];
  const float ib2 = 1.0f / (bw * bw);
  const float c1  = ib2 * 1.44269504088896340736f;  // log2(e)/bw^2
  const float c0  = -c1;                            // exp((s-1)*ib2) = exp2(fma(s,c1,c0))

  // ---- persistent B-layout fragments of X_m (wave's own rows), hi/lo split
  // B layout (16-bit, K=32): lanes0-15 hold K=0..7,16..23 (VGPR v: pair K,K+1), lanes16-31 K=8..15,24..31
  Frag BmHi[4], BmLo[4];
  {
    const float* xm = X + ((size_t)b * NN + (mw + lm)) * DD;
    #pragma unroll
    for (int t = 0; t < 4; ++t) {
      #pragma unroll
      for (int v = 0; v < 8; ++v) {
        const int K = t * 32 + ((v >> 2) << 4) + kh * 8 + ((v & 3) << 1);
        split_pair(xm[K], xm[K + 1], BmHi[t].u[v], BmLo[t].u[v]);
      }
    }
  }

  // accumulators: Y tile 16(m) x 128(d) = 8 C-fragments, + row-sum partial
  v8f acc[8];
  #pragma unroll
  for (int i = 0; i < 8; ++i)
    #pragma unroll
    for (int e = 0; e < 8; ++e) acc[i][e] = 0.0f;
  float rs = 0.0f;

  // Per-thread staging coordinates are loop-invariant:
  //   iteration 'it' covers row = it*8 + wave, cols [lane*4, lane*4+4)
  const int srow = wave;          // + it*8
  const int scol = lane * 4;

  // ---- software-pipelined global prefetch (one j-block ahead, in registers)
  float4 pf[8];
  {
    const float* X0 = X + ((size_t)b * NN) * DD;
    #pragma unroll
    for (int it = 0; it < 8; ++it)
      pf[it] = *(const float4*)(X0 + (it * 8 + srow) * DD + scol);
  }

  for (int j = 0; j < NN; j += JBLK) {
    __syncthreads();   // previous compute done reading LDS
    // ---- convert prefetched block, store to LDS planes
    #pragma unroll
    for (int it = 0; it < 8; ++it) {
      const int row = it * 8 + srow;
      const float4 vv = pf[it];
      unsigned int h01, l01, h23, l23;
      split_pair(vv.x, vv.y, h01, l01);
      split_pair(vv.z, vv.w, h23, l23);
      // row-major planes: packed 32-bit stores (4-byte aligned: LDSTR even, scol%4==0)
      *(unsigned int*)&shHi[row * LDSTR + scol]     = h01;
      *(unsigned int*)&shHi[row * LDSTR + scol + 2] = h23;
      *(unsigned int*)&shLo[row * LDSTR + scol]     = l01;
      *(unsigned int*)&shLo[row * LDSTR + scol + 2] = l23;
      // transposed hi plane: scatter 16-bit stores
      shTHi[(scol + 0) * LDSTRT + row] = (unsigned short)(h01 & 0xFFFFu);
      shTHi[(scol + 1) * LDSTRT + row] = (unsigned short)(h01 >> 16);
      shTHi[(scol + 2) * LDSTRT + row] = (unsigned short)(h23 & 0xFFFFu);
      shTHi[(scol + 3) * LDSTRT + row] = (unsigned short)(h23 >> 16);
    }
    __syncthreads();   // staged block visible to all waves (dscnt-only wait)

    // ---- issue next block's global loads AFTER the barrier: their latency is
    // covered by the whole GEMM1/exp/GEMM2 phase; consumed next iteration.
    if (j + JBLK < NN) {
      const float* Xn = X + ((size_t)b * NN + (j + JBLK)) * DD;
      #pragma unroll
      for (int it = 0; it < 8; ++it)
        pf[it] = *(const float4*)(Xn + (it * 8 + srow) * DD + scol);
    }

    // ---- GEMM1: S^T(jn,m) = X_j . X_m^T with split-bf16 (hi*hi + hi*lo + lo*hi)
    // S^T C-layout lands exactly in GEMM2's A-fragment lanes, so each score tile
    // is exp'ed and repacked into WHi/WLo immediately (no staging array).
    Frag WHi[2], WLo[2];
    #pragma unroll
    for (int jt = 0; jt < 4; ++jt) {
      v8f s;
      #pragma unroll
      for (int e = 0; e < 8; ++e) s[e] = 0.0f;
      const int abase = (jt * 16 + lm) * LDSTR;
      #pragma unroll
      for (int t = 0; t < 4; ++t) {
        Frag AjHi, AjLo;
        #pragma unroll
        for (int v = 0; v < 8; ++v) {
          // A layout (16-bit, 16x32): K = g*16 + kh*8 + 2*(v&3) (+p), g = v>>2
          const int K = t * 32 + ((v >> 2) << 4) + kh * 8 + ((v & 3) << 1);
          AjHi.u[v] = *(const unsigned int*)&shHi[abase + K];
          AjLo.u[v] = *(const unsigned int*)&shLo[abase + K];
        }
        s = __builtin_amdgcn_wmma_f32_16x16x32_bf16(false, AjHi.v, false, BmHi[t].v, (short)0, s, false, false);
        s = __builtin_amdgcn_wmma_f32_16x16x32_bf16(false, AjHi.v, false, BmLo[t].v, (short)0, s, false, false);
        s = __builtin_amdgcn_wmma_f32_16x16x32_bf16(false, AjLo.v, false, BmHi[t].v, (short)0, s, false, false);
      }
      // w = exp((s-1)/bw^2) = exp2(fma(s, c1, c0)); exponent in [-288, 0]
      float w[8];
      #pragma unroll
      for (int r = 0; r < 8; ++r) {
        w[r] = __builtin_amdgcn_exp2f(__builtin_fmaf(s[r], c1, c0));
        rs += w[r];
      }
      // repack into GEMM2 A fragments: jt -> (kt = jt>>1, vgpr group g = jt&1)
      const int kt = jt >> 1;
      const int g4 = (jt & 1) * 4;
      #pragma unroll
      for (int q = 0; q < 4; ++q)
        split_pair(w[2 * q], w[2 * q + 1], WHi[kt].u[g4 + q], WLo[kt].u[g4 + q]);
    }

    // ---- GEMM2: acc(m,d) += w(m,jn) * X(jn,d), split on w
    #pragma unroll
    for (int dt = 0; dt < 8; ++dt) {
      const int dbase = (dt * 16 + lm) * LDSTRT;
      #pragma unroll
      for (int kt = 0; kt < 2; ++kt) {
        Frag Bd;
        #pragma unroll
        for (int v = 0; v < 8; ++v) {
          const int K = kt * 32 + ((v >> 2) << 4) + kh * 8 + ((v & 3) << 1);   // K = jn, contiguous in shTHi
          Bd.u[v] = *(const unsigned int*)&shTHi[dbase + K];
        }
        acc[dt] = __builtin_amdgcn_wmma_f32_16x16x32_bf16(false, WHi[kt].v, false, Bd.v, (short)0, acc[dt], false, false);
        acc[dt] = __builtin_amdgcn_wmma_f32_16x16x32_bf16(false, WLo[kt].v, false, Bd.v, (short)0, acc[dt], false, false);
      }
    }
  }

  // ---- rowsum: lane l holds partial for m=l&15 over its jn half; combine halves
  rs += __shfl_xor(rs, 16, 32);

  // ---- divide by rowsum (C-layout row for VGPR r is m = r + kh*8)
  #pragma unroll
  for (int r = 0; r < 8; ++r) {
    const float rsum = __shfl(rs, r + kh * 8, 32);
    const float inv = 1.0f / rsum;
    #pragma unroll
    for (int dt = 0; dt < 8; ++dt) acc[dt][r] *= inv;
  }

  // ---- L2 normalize along d (reduce across the 16 lanes of each half)
  #pragma unroll
  for (int r = 0; r < 8; ++r) {
    float nn = 0.0f;
    #pragma unroll
    for (int dt = 0; dt < 8; ++dt) nn += acc[dt][r] * acc[dt][r];
    nn += __shfl_xor(nn, 1, 32);
    nn += __shfl_xor(nn, 2, 32);
    nn += __shfl_xor(nn, 4, 32);
    nn += __shfl_xor(nn, 8, 32);
    const float inv = 1.0f / fmaxf(sqrtf(nn), 1e-12f);
    #pragma unroll
    for (int dt = 0; dt < 8; ++dt) acc[dt][r] *= inv;
  }

  // ---- store Y[b, mw + r + kh*8, dt*16 + lm]
  float* Yw = Y + ((size_t)b * NN + mw) * DD;
  #pragma unroll
  for (int r = 0; r < 8; ++r) {
    const int m = r + kh * 8;
    #pragma unroll
    for (int dt = 0; dt < 8; ++dt)
      Yw[(size_t)m * DD + dt * 16 + lm] = acc[dt][r];
  }
}

extern "C" void kernel_launch(void* const* d_in, const int* in_sizes, int n_in,
                              void* d_out, int out_size, void* d_ws, size_t ws_size,
                              hipStream_t stream) {
  const float* X  = (const float*)d_in[0];
  const float* bw = (const float*)d_in[1];
  float* Y = (float*)d_out;
  (void)in_sizes; (void)n_in; (void)out_size; (void)d_ws; (void)ws_size;
  const int grid = BATCH * (NN / ROWS_BLK);   // 256 workgroups
  gbms_fused<<<grid, 256, 0, stream>>>(X, bw, Y);
}
